// BidRnn_50826642980996
// MI455X (gfx1250) — matrix operand order repaired and checked by reading, not compiled, not run on hardware
//
#include <hip/hip_runtime.h>
#include <hip/hip_bf16.h>

typedef __attribute__((ext_vector_type(16))) __bf16 v16bf;
typedef __attribute__((ext_vector_type(8)))  __bf16 bf16x8;
typedef __attribute__((ext_vector_type(8)))  float  v8f;

#define Tlen  1024
#define PITCH 528   // h row pitch in bf16 (512 + 16 pad -> avoids LDS bank conflicts)

// ---- pack fp32 weights (row-major W[n][k], gemm does h @ W^T) into the
// bf16 WMMA-B operand tile layout:
//   dst[((kt*NT + nt)*32 + lane)*16 + e] = W[nt*16 + lane%16][kt*32 + 8*(lane/16) + (e&7) + 16*(e>>3)]
// (lane%16 carries N, intra-lane the 16 K values in table order)
__global__ void pack_b_kernel(const float* __restrict__ w, __bf16* __restrict__ dst, int ntiles)
{
    const int tid   = blockIdx.x * blockDim.x + threadIdx.x;
    const int total = ntiles * 16 * 512;
    if (tid >= total) return;
    const int e    = tid & 15;
    const int lane = (tid >> 4) & 31;
    const int idx2 = tid >> 9;          // = kt*ntiles + nt
    const int nt   = idx2 % ntiles;
    const int kt   = idx2 / ntiles;
    dst[tid] = (__bf16)w[(size_t)(nt * 16 + (lane & 15)) * 512 + (kt * 32 + (lane >> 4) * 8 + (e & 7) + (e >> 3) * 16)];
}

// A-operand (16x32 bf16) fetch from LDS h:
//  lanes 0-15:  M=lane,    K = kt*32 + {0..7}  and {16..23}
//  lanes 16-31: M=lane-16, K = kt*32 + {8..15} and {24..31}
__device__ __forceinline__ v16bf wmma_load_A(const __bf16* hrow, int kt, int lh8)
{
    const bf16x8* p0 = (const bf16x8*)(hrow + kt * 32 + lh8);
    const bf16x8* p1 = (const bf16x8*)(hrow + kt * 32 + 16 + lh8);
    union { v16bf v; bf16x8 h[2]; } u;
    u.h[0] = *p0;
    u.h[1] = *p1;
    return u.v;
}

__global__ __launch_bounds__(512)
void bidgru_kernel(const float* __restrict__ x, const unsigned char* __restrict__ mask,
                   const float* __restrict__ wf_ih, const float* __restrict__ bf_ih,
                   const float* __restrict__ bf_hh,
                   const float* __restrict__ wb_ih, const float* __restrict__ bb_ih,
                   const float* __restrict__ bb_hh,
                   const float* __restrict__ p_b1, const float* __restrict__ p_w2,
                   const float* __restrict__ p_b2,
                   const __bf16* __restrict__ packGru, const __bf16* __restrict__ packP1,
                   float* __restrict__ out)
{
    __shared__ __align__(16) __bf16 h_lds[32 * PITCH];   // hidden state, bf16, 32 rows
    __shared__ float s_bih[1536], s_bhh[1536], s_wih[1536];
    __shared__ float s_pb1[512], s_pw2[512];
    __shared__ float s_part[16 * 32];    // per-wave predictor partial sums [wave][m]
    __shared__ float s_xin[32];

    const int tid  = threadIdx.x;
    const int lane = tid & 31;
    const int w    = tid >> 5;               // wave 0..15
    const int dir  = blockIdx.x & 1;         // 0 = fwd, 1 = bwd
    const int b0   = (blockIdx.x >> 1) * 32; // batch tile of 32 rows

    const float*  w_ih = dir ? wb_ih : wf_ih;
    const float*  b_ih = dir ? bb_ih : bf_ih;
    const float*  b_hh = dir ? bb_hh : bf_hh;
    const __bf16* BW   = packGru + (size_t)dir * (1536 * 512);

    for (int i = tid; i < 1536; i += 512) { s_bih[i] = b_ih[i]; s_bhh[i] = b_hh[i]; s_wih[i] = w_ih[i]; }
    for (int i = tid; i < 512;  i += 512) { s_pb1[i] = p_b1[i]; s_pw2[i] = p_w2[i]; }
    for (int i = tid; i < 32 * PITCH; i += 512) h_lds[i] = (__bf16)0.0f;
    const float pb2 = p_b2[0];
    const int   lh8 = (lane >> 4) * 8;   // 0 or 8 (lane half)
    const int   e16 = lane & 15;
    __syncthreads();

#pragma unroll 1
    for (int t = 0; t < Tlen; ++t) {
        const int t_in = dir ? (Tlen - 1 - t) : t;

        // ---- issue x/mask loads early; latency hides under the WMMA loop
        float xv = 0.0f;
        int   mk = 0;
        if (tid < 32) {
            const size_t idx = (size_t)(b0 + tid) * Tlen + t_in;
            xv = x[idx];
            mk = mask[idx];
        }

        // ---- fused K-loop: gh = h @ w_hh^T (6 tiles) and a = h @ p_w1^T (2 tiles)
        // share every A-operand load of h.
        v8f acc[2][6]  = {};   // [mt][r0,r1,z0,z1,n0,n1]
        v8f accp[2][2] = {};   // [mt][predictor tile]
        const int tiles[6] = {2 * w, 2 * w + 1, 32 + 2 * w, 33 + 2 * w, 64 + 2 * w, 65 + 2 * w};
#pragma unroll 1
        for (int kt = 0; kt < 16; ++kt) {
            v16bf bmat[6], bp[2];
#pragma unroll
            for (int c = 0; c < 6; ++c)
                bmat[c] = *(const v16bf*)(BW + (size_t)(kt * 96 + tiles[c]) * 512 + lane * 16);
#pragma unroll
            for (int j = 0; j < 2; ++j)
                bp[j] = *(const v16bf*)(packP1 + (size_t)(kt * 32 + (2 * w + j)) * 512 + lane * 16);
            if (kt < 15) {
#pragma unroll
                for (int c = 0; c < 6; ++c)
                    __builtin_prefetch(BW + (size_t)((kt + 1) * 96 + tiles[c]) * 512 + lane * 16, 0, 1);
#pragma unroll
                for (int j = 0; j < 2; ++j)
                    __builtin_prefetch(packP1 + (size_t)((kt + 1) * 32 + (2 * w + j)) * 512 + lane * 16, 0, 1);
            }
#pragma unroll
            for (int mt = 0; mt < 2; ++mt) {
                v16bf a = wmma_load_A(h_lds + (mt * 16 + e16) * PITCH, kt, lh8);
#pragma unroll
                for (int c = 0; c < 6; ++c)
                    acc[mt][c] = __builtin_amdgcn_wmma_f32_16x16x32_bf16(
                        false, a, false, bmat[c], (short)0, acc[mt][c], false, false);
#pragma unroll
                for (int j = 0; j < 2; ++j)
                    accp[mt][j] = __builtin_amdgcn_wmma_f32_16x16x32_bf16(
                        false, a, false, bp[j], (short)0, accp[mt][j], false, false);
            }
        }

        // ---- predictor epilogue: relu(+p_b1) . p_w2, butterfly-reduce over N
#pragma unroll
        for (int mt = 0; mt < 2; ++mt) {
            float part[8];
#pragma unroll
            for (int i = 0; i < 8; ++i) part[i] = 0.0f;
#pragma unroll
            for (int j = 0; j < 2; ++j) {
                const int   n  = (2 * w + j) * 16 + e16;
                const float pb = s_pb1[n], pw = s_pw2[n];
#pragma unroll
                for (int i = 0; i < 8; ++i) {
                    float av = accp[mt][j][i] + pb;
                    av = fmaxf(av, 0.0f);
                    part[i] += av * pw;
                }
            }
            // level-major butterfly: 8 independent bpermutes in flight per level
#pragma unroll
            for (int lvl = 1; lvl <= 8; lvl <<= 1) {
                float tmp[8];
#pragma unroll
                for (int i = 0; i < 8; ++i) tmp[i] = __shfl_xor(part[i], lvl, 32);
#pragma unroll
                for (int i = 0; i < 8; ++i) part[i] += tmp[i];
            }
            if (e16 == 0) {
#pragma unroll
                for (int i = 0; i < 8; ++i)
                    s_part[w * 32 + mt * 16 + i + lh8] = part[i];
            }
        }
        __syncthreads();   // h reads done (A loads) + s_part visible

        // ---- x_in = mask ? tanh(sum + p_b2) : x
        if (tid < 32) {
            float s = 0.0f;
#pragma unroll
            for (int ww = 0; ww < 16; ++ww) s += s_part[ww * 32 + tid];
            const float pred = tanhf(s + pb2);
            s_xin[tid] = mk ? pred : xv;
        }
        __syncthreads();   // xin visible

        // ---- gates + hidden update, entirely from accumulators
#pragma unroll
        for (int mt = 0; mt < 2; ++mt) {
#pragma unroll
            for (int pair = 0; pair < 2; ++pair) {
                const int   j  = 32 * w + pair * 16 + e16;
                const float wr = s_wih[j], wz = s_wih[512 + j], wn = s_wih[1024 + j];
                const float br = s_bih[j], bz = s_bih[512 + j], bn = s_bih[1024 + j];
                const float hr = s_bhh[j], hz = s_bhh[512 + j], hnb = s_bhh[1024 + j];
#pragma unroll
                for (int i = 0; i < 8; ++i) {
                    const int   m    = mt * 16 + i + lh8;   // C/D layout: M = vgpr + 8*(lane>=16)
                    const float xi   = s_xin[m];
                    const float gr   = acc[mt][pair][i]     + hr + xi * wr + br;
                    const float gz   = acc[mt][2 + pair][i] + hz + xi * wz + bz;
                    const float hn   = acc[mt][4 + pair][i] + hnb;
                    const float gin  = xi * wn + bn;
                    const float r    = 1.0f / (1.0f + __expf(-gr));
                    const float z    = 1.0f / (1.0f + __expf(-gz));
                    const float nn   = tanhf(gin + r * hn);
                    const float hold = (float)h_lds[m * PITCH + j];
                    const float hnew = (1.0f - z) * nn + z * hold;
                    h_lds[m * PITCH + j] = (__bf16)hnew;
                    out[((size_t)(b0 + m) * Tlen + t_in) * 1024 + dir * 512 + j] = hnew;
                }
            }
        }
        __syncthreads();   // h_lds updates visible before next step's A loads
    }
}

extern "C" void kernel_launch(void* const* d_in, const int* in_sizes, int n_in,
                              void* d_out, int out_size, void* d_ws, size_t ws_size,
                              hipStream_t stream)
{
    (void)in_sizes; (void)n_in; (void)out_size; (void)ws_size;
    const float*         x     = (const float*)d_in[0];
    const unsigned char* mask  = (const unsigned char*)d_in[1];
    const float*         wf_ih = (const float*)d_in[2];
    const float*         wf_hh = (const float*)d_in[3];
    const float*         bf_ih = (const float*)d_in[4];
    const float*         bf_hh = (const float*)d_in[5];
    const float*         wb_ih = (const float*)d_in[6];
    const float*         wb_hh = (const float*)d_in[7];
    const float*         bb_ih = (const float*)d_in[8];
    const float*         bb_hh = (const float*)d_in[9];
    const float*         p_w1  = (const float*)d_in[10];
    const float*         p_b1  = (const float*)d_in[11];
    const float*         p_w2  = (const float*)d_in[12];
    const float*         p_b2  = (const float*)d_in[13];
    float*               out   = (float*)d_out;

    __bf16* packF = reinterpret_cast<__bf16*>(d_ws);       // fwd w_hh  (786432 bf16)
    __bf16* packB = packF + 786432;                        // bwd w_hh  (786432 bf16)
    __bf16* packP = packB + 786432;                        // p_w1      (262144 bf16)

    pack_b_kernel<<<(786432 + 255) / 256, 256, 0, stream>>>(wf_hh, packF, 96);
    pack_b_kernel<<<(786432 + 255) / 256, 256, 0, stream>>>(wb_hh, packB, 96);
    pack_b_kernel<<<(262144 + 255) / 256, 256, 0, stream>>>(p_w1,  packP, 32);

    bidgru_kernel<<<8, 512, 0, stream>>>(x, mask, wf_ih, bf_ih, bf_hh,
                                         wb_ih, bb_ih, bb_hh,
                                         p_b1, p_w2, p_b2,
                                         packF, packP, out);
}
